// DiscrepLearning_14113262535183
// MI455X (gfx1250) — matrix operand order repaired and checked by reading, not compiled, not run on hardware
//
#include <hip/hip_runtime.h>
#include <hip/hip_bf16.h>
#include <stdint.h>

#define B_ 64
#define N_ 1024
#define M_ 1024
#define D_ 512

typedef __attribute__((ext_vector_type(16))) __bf16 v16bf;
typedef __attribute__((ext_vector_type(8)))  float  v8f;

union AF { uint32_t u[8]; v16bf v; };

__device__ __forceinline__ uint32_t bfround_(uint32_t a) {
  return a + 0x7FFFu + ((a >> 16) & 1u);   // RNE rounding bias for f32->bf16
}
__device__ __forceinline__ uint32_t pack_bf16x2(float lo, float hi) {
  uint32_t a = __builtin_bit_cast(uint32_t, lo);
  uint32_t b = __builtin_bit_cast(uint32_t, hi);
  return (bfround_(a) >> 16) | (bfround_(b) & 0xFFFF0000u);
}
__device__ __forceinline__ uint16_t to_bf16(float f) {
  return (uint16_t)(bfround_(__builtin_bit_cast(uint32_t, f)) >> 16);
}

// ---------------- Pass 1: per-(b,d) column stats -------------------------
// w[b,d]  = rsqrt(sum_n x^2) * rsqrt(sum_m y^2) ;  cs[b,d] = sum_n x[b,n,d]
__global__ __launch_bounds__(256) void dl_stats_kernel(
    const float* __restrict__ x, const float* __restrict__ y,
    float* __restrict__ w, float* __restrict__ cs) {
  int idx = blockIdx.x * 256 + threadIdx.x;       // [0, B*D)
  int b = idx / D_;
  int d = idx - b * D_;
  const float* xp = x + (size_t)b * N_ * D_ + d;
  const float* yp = y + (size_t)b * M_ * D_ + d;
  float s2x = 0.f, sx = 0.f, s2y = 0.f;
  for (int i = 0; i < N_; ++i) {
    float xa = xp[(size_t)i * D_];
    float ya = yp[(size_t)i * D_];
    s2x = fmaf(xa, xa, s2x);
    sx += xa;
    s2y = fmaf(ya, ya, s2y);
  }
  w[idx]  = rsqrtf(s2x) * rsqrtf(s2y);
  cs[idx] = sx;
}

// ---------------- Pass 2: x -> bf16 row-major copy + bf16 transpose ------
__global__ __launch_bounds__(256) void dl_cvt_kernel(
    const float* __restrict__ x, uint32_t* __restrict__ xbf,
    uint32_t* __restrict__ xT) {
  __shared__ uint16_t tile[64][66];
  const int t  = threadIdx.x;
  const int b  = blockIdx.z;
  const int n0 = blockIdx.x * 64;
  const int d0 = blockIdx.y * 64;
  const float* xb = x + ((size_t)b * N_ + n0) * D_ + d0;
#pragma unroll
  for (int i = 0; i < 8; ++i) {
    int j  = t + 256 * i;          // dword index in 64x64 tile (pairs along d)
    int r  = j >> 5;               // local n
    int cp = j & 31;               // d pair
    float f0 = xb[(size_t)r * D_ + 2 * cp];
    float f1 = xb[(size_t)r * D_ + 2 * cp + 1];
    uint32_t p = pack_bf16x2(f0, f1);
    xbf[(size_t)(b * N_ + n0 + r) * (D_ / 2) + (d0 / 2) + cp] = p;
    tile[r][2 * cp]     = (uint16_t)(p & 0xFFFFu);
    tile[r][2 * cp + 1] = (uint16_t)(p >> 16);
  }
  __syncthreads();
#pragma unroll
  for (int i = 0; i < 8; ++i) {
    int j  = t + 256 * i;          // dword index (pairs along n)
    int dr = j >> 5;               // local d
    int np = j & 31;               // n pair
    uint32_t lo = tile[2 * np][dr];
    uint32_t h2 = tile[2 * np + 1][dr];
    xT[(size_t)(b * D_ + d0 + dr) * (N_ / 2) + (n0 / 2) + np] = lo | (h2 << 16);
  }
}

// ---------------- Pass 3: fused flash-style kernel -----------------------
// grid.x = M/16 tiles, grid.y = B. 256 threads = 8 waves.
// LDS: ys bf16 Y-tile (stride 260 dw) | sl f32 S strip (stride 1028 f32)
//      pl bf16 P strip OVERLAYS sl (live only after all sl reads) |
//      stat 16x16 f32 | zinv 16 f32
#define YS_STRIDE 260
#define S_STRIDE 1028
#define P_STRIDE 1032
#define YS_BYTES (16 * YS_STRIDE * 4)
#define S_BYTES  (16 * S_STRIDE * 4)
#define STAT_OFF (YS_BYTES + S_BYTES)
#define ZINV_OFF (STAT_OFF + 1024)
#define SMEM_TOTAL (ZINV_OFF + 64)

__global__ __launch_bounds__(256) void dl_fused_kernel(
    const float* __restrict__ y, const float* __restrict__ w,
    const float* __restrict__ cs, const uint32_t* __restrict__ xbf,
    const uint32_t* __restrict__ xT, float* __restrict__ out) {
  extern __shared__ char smem[];
  uint32_t* ys   = (uint32_t*)smem;
  float*    sl   = (float*)(smem + YS_BYTES);
  uint16_t* pl   = (uint16_t*)(smem + YS_BYTES);   // overlays sl
  float*    stat = (float*)(smem + STAT_OFF);
  float*    zinv = (float*)(smem + ZINV_OFF);

  const int t    = threadIdx.x;
  const int lane = t & 31;
  const int wave = t >> 5;
  const int c16  = lane & 15;
  const int hi   = lane >> 4;          // lane-half select (0/1)
  const int b    = blockIdx.y;
  const int m0   = blockIdx.x * 16;

  // ---- Phase 0: load Y tile, scale by w, convert to bf16 in LDS ----
  {
    const float* yb = y + ((size_t)b * M_ + m0) * D_;
    const float* wb = w + (size_t)b * D_;
#pragma unroll
    for (int i = 0; i < 16; ++i) {
      int q   = t + i * 256;           // logical dword index [0, 4096)
      int row = q >> 8;
      int cp  = q & 255;
      int col = cp * 2;
      float f0 = yb[(size_t)row * D_ + col]     * wb[col];
      float f1 = yb[(size_t)row * D_ + col + 1] * wb[col + 1];
      ys[row * YS_STRIDE + cp] = pack_bf16x2(f0, f1);
    }
  }
  __syncthreads();

  // ---- Phase 1: S = (Y*w) @ X^T ; wave owns N-columns [128w,128w+128) ----
  {
    const int n0 = wave * 128;
    const uint32_t* xbfb = xbf + (size_t)b * N_ * (D_ / 2);
    v8f acc[8] = {};
    for (int kt = 0; kt < 16; ++kt) {
      // A fragment from LDS (16-bit A 16x32 layout)
      AF a;
      int base = c16 * YS_STRIDE + kt * 16 + hi * 8;
      const uint4* pa = (const uint4*)(ys + base);
      uint4 q0 = pa[0];
      uint4 q1 = pa[1];                 // +4 dwords
      a.u[0] = q0.x; a.u[1] = q0.y; a.u[2] = q0.z; a.u[3] = q0.w;
      a.u[4] = q1.x; a.u[5] = q1.y; a.u[6] = q1.z; a.u[7] = q1.w;
      // Preload ALL 8 B-fragments (16 b128 loads in flight), then 8 WMMAs.
      AF bfj[8];
#pragma unroll
      for (int j = 0; j < 8; ++j) {
        int n = n0 + j * 16 + c16;
        const uint4* pb =
            (const uint4*)(xbfb + (size_t)n * (D_ / 2) + kt * 16 + hi * 8);
        __builtin_prefetch(pb + 4, 0, 3);   // next K-chunk of this row
        uint4 b0 = pb[0];
        uint4 b1 = pb[1];
        bfj[j].u[0] = b0.x; bfj[j].u[1] = b0.y;
        bfj[j].u[2] = b0.z; bfj[j].u[3] = b0.w;
        bfj[j].u[4] = b1.x; bfj[j].u[5] = b1.y;
        bfj[j].u[6] = b1.z; bfj[j].u[7] = b1.w;
      }
#pragma unroll
      for (int j = 0; j < 8; ++j) {
        acc[j] = __builtin_amdgcn_wmma_f32_16x16x32_bf16(
            false, a.v, false, bfj[j].v, (short)0, acc[j], false, false);
      }
    }
    // Scatter S to LDS per 16x16 f32 C/D layout (VGPR r -> M=r / M=r+8)
#pragma unroll
    for (int j = 0; j < 8; ++j) {
      int col = n0 + j * 16 + c16;
#pragma unroll
      for (int r = 0; r < 8; ++r) {
        sl[(hi * 8 + r) * S_STRIDE + col] = acc[j][r];
      }
    }
  }
  __syncthreads();

  // ---- Phase 2: rowwise softmax; emit unnormalized exp as bf16 (overlay) ----
  {
    const int row = t >> 4;            // 16 threads per row
    const int sub = t & 15;
    const float* srow = sl + row * S_STRIDE;
    float mx = -3.0e38f;
    for (int i = 0; i < 64; ++i) mx = fmaxf(mx, srow[sub + 16 * i]);
    stat[row * 16 + sub] = mx;
    __syncthreads();
    float rmax = -3.0e38f;
#pragma unroll
    for (int i = 0; i < 16; ++i) rmax = fmaxf(rmax, stat[row * 16 + i]);
    __syncthreads();                    // stat reads done before reuse
    float e[64];
    float s = 0.f;
#pragma unroll
    for (int i = 0; i < 64; ++i) {
      e[i] = __expf(srow[sub + 16 * i] - rmax);
      s += e[i];
    }
    stat[row * 16 + sub] = s;
    __syncthreads();                    // ALL f32 sl reads complete here
    float z = 0.f;
#pragma unroll
    for (int i = 0; i < 16; ++i) z += stat[row * 16 + i];
    if (sub == 0) zinv[row] = 1.0f / z;
    uint16_t* prow = pl + row * P_STRIDE;   // overlays sl region (now dead)
#pragma unroll
    for (int i = 0; i < 64; ++i) prow[sub + 16 * i] = to_bf16(e[i]);
  }
  __syncthreads();

  // ---- Phase 3: O = exp(S) @ X ; wave owns D-columns [64w,64w+64) ----
  {
    const int d0 = wave * 64;
    const uint32_t* plw = (const uint32_t*)pl;     // P strip as dwords
    v8f o[4] = {};
    for (int nt = 0; nt < 32; ++nt) {
      // A fragment from P bf16 in LDS
      AF a;
      int abase = c16 * (P_STRIDE / 2) + nt * 16 + hi * 8;
      const uint4* pa = (const uint4*)(plw + abase);
      uint4 q0 = pa[0];
      uint4 q1 = pa[1];                 // +4 dwords (= +8 bf16 elements)
      a.u[0] = q0.x; a.u[1] = q0.y; a.u[2] = q0.z; a.u[3] = q0.w;
      a.u[4] = q1.x; a.u[5] = q1.y; a.u[6] = q1.z; a.u[7] = q1.w;
      // Preload 4 B-fragments, then 4 WMMAs.
      AF bfj[4];
#pragma unroll
      for (int j = 0; j < 4; ++j) {
        int dcol = d0 + j * 16 + c16;
        const uint4* pb = (const uint4*)(xT + (size_t)(b * D_ + dcol) * (N_ / 2) +
                                         nt * 16 + hi * 8);
        __builtin_prefetch(pb + 2, 0, 3);   // 16 n (8 dwords) past this lane's load on this column-row
        uint4 b0 = pb[0];
        uint4 b1 = pb[1];
        bfj[j].u[0] = b0.x; bfj[j].u[1] = b0.y;
        bfj[j].u[2] = b0.z; bfj[j].u[3] = b0.w;
        bfj[j].u[4] = b1.x; bfj[j].u[5] = b1.y;
        bfj[j].u[6] = b1.z; bfj[j].u[7] = b1.w;
      }
#pragma unroll
      for (int j = 0; j < 4; ++j) {
        o[j] = __builtin_amdgcn_wmma_f32_16x16x32_bf16(
            false, a.v, false, bfj[j].v, (short)0, o[j], false, false);
      }
    }
    // ---- Epilogue: out = colsum - o * (1/Z) ----
    const float* csb = cs + (size_t)b * D_;
#pragma unroll
    for (int j = 0; j < 4; ++j) {
      int dcol = d0 + j * 16 + c16;
      float csv = csb[dcol];
#pragma unroll
      for (int r = 0; r < 8; ++r) {
        int mrow = hi * 8 + r;
        float val = csv - o[j][r] * zinv[mrow];
        out[((size_t)b * M_ + m0 + mrow) * D_ + dcol] = val;
      }
    }
  }
}

extern "C" void kernel_launch(void* const* d_in, const int* in_sizes, int n_in,
                              void* d_out, int out_size, void* d_ws, size_t ws_size,
                              hipStream_t stream) {
  (void)in_sizes; (void)n_in; (void)out_size; (void)ws_size;
  const float* x = (const float*)d_in[0];
  const float* y = (const float*)d_in[1];
  // Workspace layout: w (B*D f32) | cs (B*D f32) | pad to 256KB |
  //                   xbf (B*N*D bf16, 64MB) | xT (B*D*N bf16, 64MB)
  float* w  = (float*)d_ws;
  float* cs = w + (size_t)B_ * D_;
  uint32_t* xbf = (uint32_t*)((char*)d_ws + (256 * 1024));
  uint32_t* xT  = xbf + ((size_t)B_ * N_ * D_ / 2);
  float* out = (float*)d_out;

  dl_stats_kernel<<<(B_ * D_) / 256, 256, 0, stream>>>(x, y, w, cs);
  dl_cvt_kernel<<<dim3(N_ / 64, D_ / 64, B_), 256, 0, stream>>>(x, xbf, xT);

  (void)hipFuncSetAttribute(reinterpret_cast<const void*>(dl_fused_kernel),
                            hipFuncAttributeMaxDynamicSharedMemorySize,
                            (int)SMEM_TOTAL);
  dl_fused_kernel<<<dim3(M_ / 16, B_), 256, SMEM_TOTAL, stream>>>(
      y, w, cs, xbf, xT, out);
}
